// Differntial_Attention_89223650607785
// MI455X (gfx1250) — matrix operand order repaired and checked by reading, not compiled, not run on hardware
//
#include <hip/hip_runtime.h>
#include <hip/hip_bf16.h>

// ---------------------------------------------------------------------------
// Differential attention for MI455X (gfx1250).
// All matmuls via v_wmma_f32_16x16x32_f16; LDS staging via
// GLOBAL_LOAD_ASYNC_TO_LDS_B128 (ASYNCcnt) with double buffering when the
// toolchain exposes the builtin.
// ---------------------------------------------------------------------------

typedef __attribute__((ext_vector_type(16))) _Float16 v16h;
typedef __attribute__((ext_vector_type(8)))  _Float16 v8h;
typedef __attribute__((ext_vector_type(8)))  float    v8f;
typedef int v4i_vs __attribute__((vector_size(4 * sizeof(int))));

#define WMMA_F16(a, b, c) \
  __builtin_amdgcn_wmma_f32_16x16x32_f16(false, (a), false, (b), (short)0, (c), false, false)

#define ATTN_SCALE 0.08838834764831845f   // 1/sqrt(128)
#define RMS_EPS    1.1920928955078125e-07f
#define LAM_INIT   0.8f

// ---- async memory->LDS staging (CDNA5), with sync fallback ---------------
#if defined(__has_builtin)
#  if __has_builtin(__builtin_amdgcn_global_load_async_to_lds_b128)
#    define USE_ASYNC_LDS 1
#  endif
#endif
#ifndef USE_ASYNC_LDS
#  define USE_ASYNC_LDS 0
#endif

typedef v4i_vs __attribute__((address_space(1)))* gptr16_t;
typedef v4i_vs __attribute__((address_space(3)))* lptr16_t;

__device__ __forceinline__ void stage16(_Float16* lds_dst, const _Float16* gsrc) {
#if USE_ASYNC_LDS
  __builtin_amdgcn_global_load_async_to_lds_b128((gptr16_t)(gsrc), (lptr16_t)(lds_dst), 0, 0);
#else
  *(v8h*)lds_dst = *(const v8h*)gsrc;
#endif
}

__device__ __forceinline__ void stage_fence() {
#if USE_ASYNC_LDS
#  if defined(__has_builtin) && __has_builtin(__builtin_amdgcn_s_wait_asynccnt)
  __builtin_amdgcn_s_wait_asynccnt(0);
#  else
  asm volatile("s_wait_asynccnt 0x0" ::: "memory");
#  endif
#endif
}

// A-fragment (16 M x 32 K, f16) per ISA 7.12.2: lane holds row (lane&15);
// halves 0-7 at K = koff+{0..7}, halves 8-15 at K = koff+16+{0..7}, koff = (lane<16)?0:8.
__device__ __forceinline__ v16h frag_a(const _Float16* base, int row, int ld, int kk, int lane) {
  const _Float16* p = base + (size_t)(row + (lane & 15)) * ld + kk + ((lane & 16) ? 8 : 0);
  v8h lo = *(const v8h*)p;
  v8h hi = *(const v8h*)(p + 16);
  return __builtin_shufflevector(lo, hi, 0,1,2,3,4,5,6,7,8,9,10,11,12,13,14,15);
}

// B-fragment (32 K x 16 N, f16): lane holds column (lane&15); same K order as A:
// halves 0-7 at K = koff+{0..7}, halves 8-15 at K = koff+16+{0..7}, koff = (lane<16)?0:8.
// Source row-major (N, K) -> two 16-byte loads per lane.
__device__ __forceinline__ v16h frag_b(const _Float16* base, int col, int ld, int kk, int lane) {
  const _Float16* p = base + (size_t)(col + (lane & 15)) * ld + kk + ((lane & 16) ? 8 : 0);
  v8h lo = *(const v8h*)p;
  v8h hi = *(const v8h*)(p + 16);
  return __builtin_shufflevector(lo, hi, 0,1,2,3,4,5,6,7,8,9,10,11,12,13,14,15);
}

// ---------------------------------------------------------------------------
// fp32 -> f16 convert
// ---------------------------------------------------------------------------
__global__ void cvt_f16_kernel(const float* __restrict__ in, _Float16* __restrict__ out, int n) {
  int i = blockIdx.x * blockDim.x + threadIdx.x;
  if (i < n) out[i] = (_Float16)in[i];
}

// ---------------------------------------------------------------------------
// lambda = exp(<lq1,lk1>) - exp(<lq2,lk2>) + 0.8
// ---------------------------------------------------------------------------
__global__ void lam_kernel(const float* __restrict__ lq1, const float* __restrict__ lq2,
                           const float* __restrict__ lk1, const float* __restrict__ lk2,
                           float* __restrict__ lam) {
  if (threadIdx.x == 0) {
    float d1 = 0.f, d2 = 0.f;
    for (int i = 0; i < 128; ++i) { d1 += lq1[i] * lk1[i]; d2 += lq2[i] * lk2[i]; }
    *lam = __expf(d1) - __expf(d2) + LAM_INIT;
  }
}

// ---------------------------------------------------------------------------
// C(M,N) = A(M,K) * W(N,K)^T ; A,W f16, C f16 or f32.
// 128x128 block tile, 8 waves of 64x32, double-buffered 64-wide K panels
// staged with async memory->LDS transfers overlapping the WMMA stream.
// ---------------------------------------------------------------------------
template <typename OutT>
__global__ __launch_bounds__(256) void gemm_nt_kernel(
    const _Float16* __restrict__ A, const _Float16* __restrict__ W,
    OutT* __restrict__ C, int M, int N, int Kdim) {
  __shared__ _Float16 As[2][128][72];
  __shared__ _Float16 Ws[2][128][72];

  const int tid  = threadIdx.x;
  const int lane = tid & 31;
  const int w    = tid >> 5;
  const int bm   = blockIdx.y * 128;
  const int bn   = blockIdx.x * 128;
  const int wm   = (w & 1) * 64;
  const int wn   = (w >> 1) * 32;

  // per-thread staging coordinates: 4 x 16B for A, 4 x 16B for W per panel
  int srow[4], scol[4];
#pragma unroll
  for (int i = 0; i < 4; ++i) {
    int li  = tid + 256 * i;
    srow[i] = li >> 3;
    scol[i] = (li & 7) << 3;
  }

  v8f acc[4][2] = {};

  // prologue: stage panel 0 into buffer 0
#pragma unroll
  for (int i = 0; i < 4; ++i) {
    stage16(&As[0][srow[i]][scol[i]], &A[(size_t)(bm + srow[i]) * Kdim + scol[i]]);
    stage16(&Ws[0][srow[i]][scol[i]], &W[(size_t)(bn + srow[i]) * Kdim + scol[i]]);
  }
  stage_fence();
  __syncthreads();

  int cur = 0;
  for (int k0 = 0; k0 < Kdim; k0 += 64) {
    const int nxt = cur ^ 1;
    if (k0 + 64 < Kdim) {  // prefetch next panel while computing this one
#pragma unroll
      for (int i = 0; i < 4; ++i) {
        stage16(&As[nxt][srow[i]][scol[i]], &A[(size_t)(bm + srow[i]) * Kdim + k0 + 64 + scol[i]]);
        stage16(&Ws[nxt][srow[i]][scol[i]], &W[(size_t)(bn + srow[i]) * Kdim + k0 + 64 + scol[i]]);
      }
    }
#pragma unroll
    for (int kk = 0; kk < 64; kk += 32) {
      v16h af[4], bf[2];
#pragma unroll
      for (int i = 0; i < 4; ++i) af[i] = frag_a(&As[cur][0][0], wm + i * 16, 72, kk, lane);
#pragma unroll
      for (int j = 0; j < 2; ++j) bf[j] = frag_b(&Ws[cur][0][0], wn + j * 16, 72, kk, lane);
#pragma unroll
      for (int i = 0; i < 4; ++i)
#pragma unroll
        for (int j = 0; j < 2; ++j)
          acc[i][j] = WMMA_F16(af[i], bf[j], acc[i][j]);
    }
    stage_fence();
    __syncthreads();
    cur = nxt;
  }

  const int hi8 = (lane & 16) ? 8 : 0;
  const int nl  = lane & 15;
#pragma unroll
  for (int i = 0; i < 4; ++i)
#pragma unroll
    for (int j = 0; j < 2; ++j)
#pragma unroll
      for (int r = 0; r < 8; ++r) {
        int m = bm + wm + i * 16 + r + hi8;
        int n = bn + wn + j * 16 + nl;
        if (m < M && n < N) C[(size_t)m * N + n] = (OutT)acc[i][j][r];
      }
}

// ---------------------------------------------------------------------------
// Dual-stream flash attention + RMSNorm + scrambled scatter into T.
// Q/K/V: (4096, 2048) f16, rows = b*2048 + s, cols = h*128 + e.
// Block = 64 queries of one (b,h); 8 waves:
//   wave w: qb = w&3 (16 queries), st = w>>2 (score stream AND owned e-half).
// T layout (torch-faithful scramble): T[b][e*16 + h][s]  (f16)
// ---------------------------------------------------------------------------
__global__ __launch_bounds__(256) void diff_attn_kernel(
    const _Float16* __restrict__ Q, const _Float16* __restrict__ Kmat,
    const _Float16* __restrict__ V, const float* __restrict__ lam_p,
    _Float16* __restrict__ T) {
  __shared__ _Float16 Qs[64][136];
  __shared__ _Float16 Ks[64][136];
  __shared__ _Float16 Vt[128][72];      // transposed V tile: [e][key]
  __shared__ _Float16 Pb[2][64][72];    // exp(P) per stream: [q][key]
  __shared__ float alpha_s[2][64];
  __shared__ float lsum_s[2][64];
  __shared__ float rms_s[2][64];

  const int tid  = threadIdx.x;
  const int lane = tid & 31;
  const int w    = tid >> 5;
  const int qb   = w & 3;
  const int st   = w >> 2;
  const int eoff = st * 64;
  const int qt   = blockIdx.x;   // query tile 0..31
  const int h    = blockIdx.y;   // head 0..15
  const int b    = blockIdx.z;   // batch 0..1
  const int D    = 2048;
  const int hi8  = (lane & 16) ? 8 : 0;
  const int nl   = lane & 15;

  const float lamv = *lam_p;
  const size_t rowQ = (size_t)(b * 2048 + qt * 64);

  // stage the Q tile once (async; fenced before the first barrier below)
#pragma unroll
  for (int i = 0; i < 4; ++i) {
    int li  = tid + 256 * i;
    int row = li >> 4;
    int c8  = (li & 15) << 3;
    stage16(&Qs[row][c8], &Q[(rowQ + row) * D + h * 128 + c8]);
  }

  v8f O[2][4] = {};             // [stream][e-frag] over owned 64-wide e-half
  float mrow[8], lrow[8];
#pragma unroll
  for (int r = 0; r < 8; ++r) { mrow[r] = -1e30f; lrow[r] = 0.f; }

  for (int kt = 0; kt < 32; ++kt) {
    const size_t rowK = (size_t)(b * 2048 + kt * 64);
#pragma unroll
    for (int i = 0; i < 4; ++i) {
      int li  = tid + 256 * i;
      int row = li >> 4;
      int c8  = (li & 15) << 3;
      stage16(&Ks[row][c8], &Kmat[(rowK + row) * D + h * 128 + c8]);
      v8h vv = *(const v8h*)&V[(rowK + row) * D + h * 128 + c8];
#pragma unroll
      for (int j = 0; j < 8; ++j) Vt[c8 + j][row] = vv[j];   // transpose into LDS
    }
    stage_fence();
    __syncthreads();

    // ---- scores S = Q_st * K_st^T for this wave's 16 queries x 64 keys ----
    v8f S[4] = {};
#pragma unroll
    for (int ks = 0; ks < 64; ks += 32) {
      v16h aq = frag_a(&Qs[0][0], qb * 16, 136, st * 64 + ks, lane);
#pragma unroll
      for (int nf = 0; nf < 4; ++nf) {
        v16h bk = frag_b(&Ks[0][0], nf * 16, 136, st * 64 + ks, lane);
        S[nf] = WMMA_F16(aq, bk, S[nf]);
      }
    }

    // ---- online softmax update, publish exp(P) and alpha ----
#pragma unroll
    for (int r = 0; r < 8; ++r) {
      float mx = -1e30f;
#pragma unroll
      for (int nf = 0; nf < 4; ++nf) {
        float v = S[nf][r] * ATTN_SCALE;
        S[nf][r] = v;
        mx = fmaxf(mx, v);
      }
#pragma unroll
      for (int m2 = 1; m2 < 16; m2 <<= 1) mx = fmaxf(mx, __shfl_xor(mx, m2, 32));
      float mnew  = fmaxf(mrow[r], mx);
      float alpha = __expf(mrow[r] - mnew);
      float rs = 0.f;
#pragma unroll
      for (int nf = 0; nf < 4; ++nf) {
        float p = __expf(S[nf][r] - mnew);
        S[nf][r] = p;
        rs += p;
      }
#pragma unroll
      for (int m2 = 1; m2 < 16; m2 <<= 1) rs += __shfl_xor(rs, m2, 32);
      lrow[r] = lrow[r] * alpha + rs;
      mrow[r] = mnew;
      const int rowi = qb * 16 + r + hi8;
      if (nl == 0) alpha_s[st][rowi] = alpha;
#pragma unroll
      for (int nf = 0; nf < 4; ++nf) Pb[st][rowi][nf * 16 + nl] = (_Float16)S[nf][r];
    }
    __syncthreads();

    // ---- rescale accumulators and accumulate P*V for both streams ----
#pragma unroll
    for (int t = 0; t < 2; ++t) {
#pragma unroll
      for (int r = 0; r < 8; ++r) {
        float af = alpha_s[t][qb * 16 + r + hi8];
#pragma unroll
        for (int nf = 0; nf < 4; ++nf) O[t][nf][r] *= af;
      }
#pragma unroll
      for (int ks = 0; ks < 64; ks += 32) {
        v16h ap = frag_a(&Pb[t][0][0], qb * 16, 72, ks, lane);
#pragma unroll
        for (int nf = 0; nf < 4; ++nf) {
          v16h bv = frag_b(&Vt[0][0], eoff + nf * 16, 72, ks, lane);
          O[t][nf] = WMMA_F16(ap, bv, O[t][nf]);
        }
      }
    }
    __syncthreads();
  }

  // ---- combine streams, RMSNorm over full head (cross-wave), scatter to T ----
#pragma unroll
  for (int r = 0; r < 8; ++r)
    if (nl == 0) lsum_s[st][qb * 16 + r + hi8] = lrow[r];
  __syncthreads();

#pragma unroll
  for (int r = 0; r < 8; ++r) {
    const int rowi = qb * 16 + r + hi8;
    float inv1 = 1.f / lsum_s[0][rowi];
    float inv2 = 1.f / lsum_s[1][rowi];
    float ss = 0.f;
#pragma unroll
    for (int nf = 0; nf < 4; ++nf) {
      float v = O[0][nf][r] * inv1 - lamv * (O[1][nf][r] * inv2);
      O[0][nf][r] = v;
      ss += v * v;
    }
#pragma unroll
    for (int m2 = 1; m2 < 16; m2 <<= 1) ss += __shfl_xor(ss, m2, 32);
    if (nl == 0) rms_s[st][rowi] = ss;
  }
  __syncthreads();

  float scl[8];
#pragma unroll
  for (int r = 0; r < 8; ++r) {
    const int rowi = qb * 16 + r + hi8;
    float mean = (rms_s[0][rowi] + rms_s[1][rowi]) * (1.f / 128.f);
    scl[r] = (1.f - LAM_INIT) * rsqrtf(mean + RMS_EPS);
  }

  const int sbase = qt * 64;
#pragma unroll
  for (int nf = 0; nf < 4; ++nf) {
    int e = eoff + nf * 16 + nl;
    size_t tb = ((size_t)b * 2048 + (size_t)e * 16 + h) * 2048;
#pragma unroll
    for (int r = 0; r < 8; ++r) {
      int srow = sbase + qb * 16 + r + hi8;
      T[tb + srow] = (_Float16)(O[0][nf][r] * scl[r]);
    }
  }
}

// ---------------------------------------------------------------------------
// launch
// ---------------------------------------------------------------------------
extern "C" void kernel_launch(void* const* d_in, const int* in_sizes, int n_in,
                              void* d_out, int out_size, void* d_ws, size_t ws_size,
                              hipStream_t stream) {
  const float* x   = (const float*)d_in[0];
  const float* wq  = (const float*)d_in[1];
  const float* wk  = (const float*)d_in[2];
  const float* wv  = (const float*)d_in[3];
  const float* wo  = (const float*)d_in[4];
  const float* lq1 = (const float*)d_in[5];
  const float* lq2 = (const float*)d_in[6];
  const float* lk1 = (const float*)d_in[7];
  const float* lk2 = (const float*)d_in[8];

  char* ws = (char*)d_ws;
  const size_t MB = (size_t)1 << 20;
  _Float16* xh  = (_Float16*)(ws + 0 * MB);    // 16 MB
  _Float16* wqh = (_Float16*)(ws + 16 * MB);   //  8 MB
  _Float16* wkh = (_Float16*)(ws + 24 * MB);
  _Float16* wvh = (_Float16*)(ws + 32 * MB);
  _Float16* woh = (_Float16*)(ws + 40 * MB);
  _Float16* Qh  = (_Float16*)(ws + 48 * MB);   // 16 MB each
  _Float16* Kh  = (_Float16*)(ws + 64 * MB);
  _Float16* Vh  = (_Float16*)(ws + 80 * MB);
  _Float16* Th  = (_Float16*)(ws + 96 * MB);
  float*    lam = (float*)(ws + 112 * MB);

  const int nx = 2 * 2048 * 2048;
  const int nw = 2048 * 2048;
  cvt_f16_kernel<<<nx / 256, 256, 0, stream>>>(x, xh, nx);
  cvt_f16_kernel<<<nw / 256, 256, 0, stream>>>(wq, wqh, nw);
  cvt_f16_kernel<<<nw / 256, 256, 0, stream>>>(wk, wkh, nw);
  cvt_f16_kernel<<<nw / 256, 256, 0, stream>>>(wv, wvh, nw);
  cvt_f16_kernel<<<nw / 256, 256, 0, stream>>>(wo, woh, nw);
  lam_kernel<<<1, 32, 0, stream>>>(lq1, lq2, lk1, lk2, lam);

  dim3 gg(16, 32);  // N/128, M/128
  gemm_nt_kernel<_Float16><<<gg, 256, 0, stream>>>(xh, wqh, Qh, 4096, 2048, 2048);
  gemm_nt_kernel<_Float16><<<gg, 256, 0, stream>>>(xh, wkh, Kh, 4096, 2048, 2048);
  gemm_nt_kernel<_Float16><<<gg, 256, 0, stream>>>(xh, wvh, Vh, 4096, 2048, 2048);

  dim3 ga(32, 16, 2);  // query tiles, heads, batch
  diff_attn_kernel<<<ga, 256, 0, stream>>>(Qh, Kh, Vh, lam, Th);

  gemm_nt_kernel<float><<<gg, 256, 0, stream>>>(Th, woh, (float*)d_out, 4096, 2048, 2048);
}
